// Model_24730421690435
// MI455X (gfx1250) — matrix operand order repaired and checked by reading, not compiled, not run on hardware
//
#include <hip/hip_runtime.h>
#include <stdint.h>

// ---------------------------------------------------------------------------
// NTM-style recurrent model, persistent-kernel formulation for MI455X (gfx1250)
//   B=2048 batch rows -> 128 workgroups x 16 rows, 33 sequential steps each.
//   GEMMs via v_wmma_f32_16x16x32_bf16 (f32 accumulate), activations + LSTM
//   cell state staged in LDS, NTM memory in L2-resident workspace.
//   Round 3: zero-init accumulators (bias in epilogue), LSTM c-state in LDS,
//   compact (non-unrolled) tile loop -- kills the scratch spills seen in r2.
// ---------------------------------------------------------------------------

constexpr int kB  = 2048;
constexpr int kT  = 32;     // scan runs t = 0..32 (33 steps)
constexpr int kZ  = 128;
constexpr int kM  = 256;
constexpr int kH  = 512;
constexpr int kN  = 10;
constexpr int kZM = kZ + kM;        // 384  (lstm_in width)
constexpr int kKA = kZM + kH;       // 896  (fused gate-GEMM K)
constexpr int kG  = 4 * kH;         // 2048 (gate outputs)
constexpr int kHO = 1056;           // head outputs padded 1040 -> 66*16
constexpr int BT  = 16;             // batch tile per workgroup
constexpr int KC_G = kKA / 32;      // 28 k-chunks, gate GEMM
constexpr int KC_H = kH / 32;       // 16 k-chunks, head GEMM

// LDS layout (bytes)
constexpr int ASTR   = 904;                 // A-tile stride in ushorts (bank-spread pad)
constexpr int HSTR   = 1072;                // heads stride in floats
constexpr int A_BY   = BT * ASTR * 2;       // 28928
constexpr int H_OFF  = A_BY;
constexpr int RV_OFF = H_OFF + BT * HSTR * 4;     // 97536
constexpr int WR_OFF = RV_OFF + BT * kM * 4;      // 113920
constexpr int WW_OFF = WR_OFF + BT * kN * 4;
constexpr int WN_OFF = WW_OFF + BT * kN * 4;
constexpr int DR_OFF = WN_OFF + BT * kN * 4;
constexpr int DW_OFF = DR_OFF + BT * kN * 4;
constexpr int MN_OFF = DW_OFF + BT * kN * 4;
constexpr int KR_OFF = MN_OFF + BT * kN * 4;
constexpr int KW_OFF = KR_OFF + BT * 4;
constexpr int C_OFF  = KW_OFF + BT * 4;           // LSTM cell state [BT][kH] f32
constexpr int SMEM_BYTES = C_OFF + BT * kH * 4;   // ~150.7 KB dynamic LDS

// Workspace layout (bytes)
constexpr size_t OFF_WA  = 0;                                  // 2048*896 bf16
constexpr size_t OFF_WH  = OFF_WA + (size_t)kG * kKA * 2;      // 1056*512 bf16
constexpr size_t OFF_BG  = OFF_WH + (size_t)kHO * kH * 2;      // 2048 f32
constexpr size_t OFF_BH  = OFF_BG + (size_t)kG * 4;            // 1056 f32
constexpr size_t OFF_ZN  = OFF_BH + 4352;                      // 2048*33*128 bf16
constexpr size_t OFF_MEM = OFF_ZN + (size_t)kB * 33 * kZ * 2;  // 2048*10*256 f32

typedef __bf16 v16bf __attribute__((ext_vector_type(16)));
typedef float  v8f   __attribute__((ext_vector_type(8)));

union Frag { v16bf v; uint32_t d[8]; };

__device__ __forceinline__ unsigned short f2bf(float x) {
    union { float f; uint32_t u; } v; v.f = x;
    uint32_t r = v.u + 0x7FFFu + ((v.u >> 16) & 1u);   // round-to-nearest-even
    return (unsigned short)(r >> 16);
}
__device__ __forceinline__ float sigf(float x) { return 1.f / (1.f + __expf(-x)); }
__device__ __forceinline__ float softplusf(float x) { return x > 20.f ? x : log1pf(__expf(x)); }
// single-transcendental tanh: tanh(|x|) = (1 - e^{-2|x|}) / (1 + e^{-2|x|})
__device__ __forceinline__ float tanhf_fast(float x) {
    float e = __expf(-2.f * fabsf(x));
    float t = (1.f - e) / (1.f + e);
    return copysignf(t, x);
}

// 16-bit A fragment K offsets (ISA 7.12.2): V0..3 -> K 0..7 / 8..15, V4..7 -> K 16..23 / 24..31
__device__ __forceinline__ int kkA(int r, int hi) { return 2 * r + (r >= 4 ? 8 : 0) + 8 * hi; }
// B fragment: same K order as A: V0..3 -> K 0..7 / 8..15, V4..7 -> K 16..23 / 24..31
__device__ __forceinline__ int kkB(int r, int hi) { return 2 * r + (r >= 4 ? 8 : 0) + 8 * hi; }

__device__ __forceinline__ v8f wmma_bf16(Frag a, Frag b, v8f c) {
    return __builtin_amdgcn_wmma_f32_16x16x32_bf16(false, a.v, false, b.v, (short)0, c, false, false);
}

// ------------------------------- prep kernels ------------------------------

__global__ void pack_gate_weights(const float* __restrict__ W_ih, const float* __restrict__ W_hh,
                                  const float* __restrict__ b_ih, const float* __restrict__ b_hh,
                                  unsigned short* __restrict__ Wa, float* __restrict__ bias_g) {
    int idx = blockIdx.x * blockDim.x + threadIdx.x;
    if (idx < kG * kKA) {
        int o = idx / kKA, k = idx - o * kKA;
        float v = (k < kZM) ? W_ih[o * kZM + k] : W_hh[o * kH + (k - kZM)];
        Wa[idx] = f2bf(v);
    }
    if (idx < kG) bias_g[idx] = b_ih[idx] + b_hh[idx];
}

struct HeadPtrs { const float* W[13]; const float* b[13]; };

__global__ void pack_head_weights(HeadPtrs hp, unsigned short* __restrict__ Wh,
                                  float* __restrict__ bias_h) {
    const int hs[13] = {256, 256, 256, 256, 4, 1, 1, 1, 1, 3, 3, 1, 1}; // setup_inputs order
    int idx = blockIdx.x * blockDim.x + threadIdx.x;
    if (idx < kHO * kH) {
        int o = idx >> 9, k = idx & 511;
        float v = 0.f;
        if (o < 1040) {
            int off = 0;
            for (int j = 0; j < 13; ++j) {
                if (o < off + hs[j]) { v = hp.W[j][(o - off) * kH + k]; break; }
                off += hs[j];
            }
        }
        Wh[idx] = f2bf(v);
    }
    if (idx < kHO) {
        float v = 0.f;
        if (idx < 1040) {
            int off = 0;
            for (int j = 0; j < 13; ++j) {
                if (idx < off + hs[j]) { v = hp.b[j][idx - off]; break; }
                off += hs[j];
            }
        }
        bias_h[idx] = v;
    }
}

// LayerNorm over time axis (ddof=1), cast to bf16, append zero step at t=32.
__global__ void layernorm_cast(const float* __restrict__ z_seq, const float* __restrict__ gamma,
                               const float* __restrict__ beta, unsigned short* __restrict__ zn) {
    int idx = blockIdx.x * blockDim.x + threadIdx.x;
    if (idx >= kB * kZ) return;
    int b = idx >> 7, f = idx & 127;
    float xv[kT];
    float s = 0.f, s2 = 0.f;
    #pragma unroll 4
    for (int t = 0; t < kT; ++t) {
        float x = z_seq[(size_t)b * kT * kZ + t * kZ + f];
        xv[t] = x; s += x; s2 += x * x;
    }
    float mu  = s * (1.f / kT);
    float var = (s2 - kT * mu * mu) * (1.f / (kT - 1));
    float inv = 1.f / sqrtf(var + 1e-8f);
    float g = gamma[f], be = beta[f];
    for (int t = 0; t < kT; ++t)
        zn[(size_t)b * 33 * kZ + t * kZ + f] = f2bf((xv[t] - mu) * inv * g + be);
    zn[(size_t)b * 33 * kZ + kT * kZ + f] = 0;
}

// --------------------------- addressing helper -----------------------------

__device__ void address10(const float* dt, float kn, const float* mn, float strength,
                          float gate, const float* sh, float sharpen,
                          const float* wprev, int use_interp, float* wout) {
    float e[kN], mx = -1e30f;
    #pragma unroll
    for (int n = 0; n < kN; ++n) {
        float cs = dt[n] / (kn * fmaxf(mn[n], 1e-8f));
        e[n] = strength * cs;
        mx = fmaxf(mx, e[n]);
    }
    float s = 0.f;
    #pragma unroll
    for (int n = 0; n < kN; ++n) { e[n] = __expf(e[n] - mx); s += e[n]; }
    float w[kN];
    float is = 1.f / s;
    #pragma unroll
    for (int n = 0; n < kN; ++n) w[n] = e[n] * is;
    if (use_interp) {
        #pragma unroll
        for (int n = 0; n < kN; ++n) w[n] = gate * w[n] + (1.f - gate) * wprev[n];
    }
    float w2[kN];
    #pragma unroll
    for (int n = 0; n < kN; ++n)
        w2[n] = sh[0] * w[(n + 1) % kN] + sh[1] * w[n] + sh[2] * w[(n + kN - 1) % kN];
    float s2 = 0.f;
    #pragma unroll
    for (int n = 0; n < kN; ++n) { float p = powf(fmaxf(w2[n], 0.f), sharpen); wout[n] = p; s2 += p; }
    float r = 1.f / s2;
    #pragma unroll
    for (int n = 0; n < kN; ++n) wout[n] *= r;
}

// ----------------------------- persistent kernel ---------------------------

__global__ void ntm_persistent(const unsigned short* __restrict__ Wa,
                               const unsigned short* __restrict__ Wh,
                               const float* __restrict__ bias_g,
                               const float* __restrict__ bias_h,
                               const unsigned short* __restrict__ znorm,
                               float* __restrict__ memws,
                               const float* __restrict__ memory_init,
                               float* __restrict__ out) {
    extern __shared__ unsigned char smem[];
    unsigned short* A_ls = (unsigned short*)smem;            // [BT][ASTR] bf16: z|rv|h
    float* heads = (float*)(smem + H_OFF);                   // [BT][HSTR]
    float* rvec  = (float*)(smem + RV_OFF);                  // [BT][256]
    float* wr_ls = (float*)(smem + WR_OFF);                  // [BT][10]
    float* ww_ls = (float*)(smem + WW_OFF);
    float* wn_ls = (float*)(smem + WN_OFF);
    float* drs   = (float*)(smem + DR_OFF);
    float* dws   = (float*)(smem + DW_OFF);
    float* mns   = (float*)(smem + MN_OFF);
    float* knr   = (float*)(smem + KR_OFF);
    float* knw   = (float*)(smem + KW_OFF);
    float* c_ls  = (float*)(smem + C_OFF);                   // [BT][kH] LSTM cell

    const int tid  = threadIdx.x;
    const int lane = tid & 31;
    const int wv   = tid >> 5;         // 8 waves
    const int nn   = lane & 15;
    const int hi   = lane >> 4;
    const int b0   = blockIdx.x * BT;

    // ---- init: h cols, c state, read_vec, address weights, NTM memory ----
    for (int i = tid; i < BT * kH; i += 256) {
        A_ls[(i >> 9) * ASTR + kZM + (i & 511)] = 0;
        c_ls[i] = 0.f;
    }
    for (int i = tid; i < BT * kM; i += 256) rvec[i] = 0.f;
    for (int i = tid; i < BT * kN; i += 256) { wr_ls[i] = 0.f; ww_ls[i] = 0.f; wn_ls[i] = 0.f; }
    for (int i = tid; i < BT * kN * kM; i += 256) {
        int b = i / (kN * kM), rest = i - b * (kN * kM);
        memws[(size_t)(b0 + b) * kN * kM + rest] = memory_init[rest];
    }
    __syncthreads();

    for (int t = 0; t <= kT; ++t) {
        // ---- Phase A: stage z_t and read_vec into A-tile (bf16) ----
        for (int i = tid; i < BT * kZ; i += 256) {
            int row = i >> 7, f = i & 127;
            A_ls[row * ASTR + f] = znorm[(size_t)(b0 + row) * 33 * kZ + t * kZ + f];
        }
        for (int i = tid; i < BT * kM; i += 256) {
            int row = i >> 8, m = i & 255;
            A_ls[row * ASTR + kZ + m] = f2bf(rvec[i]);
        }
        __syncthreads();

        // ---- Phase B: gate GEMM, software-pipelined operand fragments ----
        #pragma unroll 1
        for (int ul = 0; ul < 4; ++ul) {
            const int u = wv * 4 + ul;                 // hidden-unit tile (16 units)
            const unsigned short* wr4[4];
            #pragma unroll
            for (int g = 0; g < 4; ++g)
                wr4[g] = Wa + (size_t)(g * kH + u * 16 + nn) * kKA;

            v8f acc[4];
            #pragma unroll
            for (int g = 0; g < 4; ++g)
                #pragma unroll
                for (int r = 0; r < 8; ++r) acc[g][r] = 0.f;   // bias applied in epilogue

            Frag a0, b0[4];
            #pragma unroll
            for (int r = 0; r < 8; ++r)
                a0.d[r] = *(const uint32_t*)(&A_ls[nn * ASTR + kkA(r, hi)]);
            #pragma unroll
            for (int g = 0; g < 4; ++g)
                #pragma unroll
                for (int r = 0; r < 8; ++r)
                    b0[g].d[r] = *(const uint32_t*)(wr4[g] + kkB(r, hi));

            for (int kc = 0; kc < KC_G; ++kc) {
                Frag a1, b1[4];
                if (kc + 1 < KC_G) {                 // prefetch next k-chunk
                    const int K1 = (kc + 1) * 32;
                    #pragma unroll
                    for (int r = 0; r < 8; ++r)
                        a1.d[r] = *(const uint32_t*)(&A_ls[nn * ASTR + K1 + kkA(r, hi)]);
                    #pragma unroll
                    for (int g = 0; g < 4; ++g)
                        #pragma unroll
                        for (int r = 0; r < 8; ++r)
                            b1[g].d[r] = *(const uint32_t*)(wr4[g] + K1 + kkB(r, hi));
                }
                #pragma unroll
                for (int g = 0; g < 4; ++g)
                    acc[g] = wmma_bf16(a0, b0[g], acc[g]);
                a0 = a1;
                #pragma unroll
                for (int g = 0; g < 4; ++g) b0[g] = b1[g];
            }

            // epilogue: bias + LSTM cell update (c lives in LDS, own slots only)
            const int unit = u * 16 + nn;
            float bgi = bias_g[0 * kH + unit];
            float bgf = bias_g[1 * kH + unit];
            float bgg = bias_g[2 * kH + unit];
            float bgo = bias_g[3 * kH + unit];
            #pragma unroll
            for (int r = 0; r < 8; ++r) {
                const int m = r + 8 * hi;
                float iv = acc[0][r] + bgi, fv = acc[1][r] + bgf;
                float gv = acc[2][r] + bgg, ov = acc[3][r] + bgo;
                float c = sigf(fv) * c_ls[m * kH + unit] + sigf(iv) * tanhf_fast(gv);
                c_ls[m * kH + unit] = c;
                float h = sigf(ov) * tanhf_fast(c);
                A_ls[m * ASTR + kZM + unit] = f2bf(h);
            }
        }
        __syncthreads();

        // ---- Phase C: head GEMM  heads = h @ Wh.T + bias (pipelined) ----
        #pragma unroll 1
        for (int nt = wv; nt < kHO / 16; nt += 8) {
            const int o0 = nt * 16;
            const unsigned short* wrow = Wh + (size_t)(o0 + nn) * kH;
            v8f acc;
            #pragma unroll
            for (int r = 0; r < 8; ++r) acc[r] = 0.f;

            Frag a0, b0;
            #pragma unroll
            for (int r = 0; r < 8; ++r) {
                a0.d[r] = *(const uint32_t*)(&A_ls[nn * ASTR + kZM + kkA(r, hi)]);
                b0.d[r] = *(const uint32_t*)(wrow + kkB(r, hi));
            }
            for (int kc = 0; kc < KC_H; ++kc) {
                Frag a1, b1;
                if (kc + 1 < KC_H) {
                    const int K1 = (kc + 1) * 32;
                    #pragma unroll
                    for (int r = 0; r < 8; ++r) {
                        a1.d[r] = *(const uint32_t*)(&A_ls[nn * ASTR + kZM + K1 + kkA(r, hi)]);
                        b1.d[r] = *(const uint32_t*)(wrow + K1 + kkB(r, hi));
                    }
                }
                acc = wmma_bf16(a0, b0, acc);
                a0 = a1; b0 = b1;
            }
            float bv = bias_h[o0 + nn];
            #pragma unroll
            for (int r = 0; r < 8; ++r)
                heads[(r + 8 * hi) * HSTR + o0 + nn] = acc[r] + bv;
        }
        __syncthreads();

        if (t == kT) {   // last step: emit y + argmax, carry is dead
            if (tid < BT) {
                float y[4];
                #pragma unroll
                for (int j = 0; j < 4; ++j) {
                    y[j] = heads[tid * HSTR + 1024 + j];
                    out[(size_t)(b0 + tid) * 4 + j] = y[j];
                }
                int bi = 0; float best = y[0];
                #pragma unroll
                for (int j = 1; j < 4; ++j) if (y[j] > best) { best = y[j]; bi = j; }
                out[(size_t)kB * 4 + b0 + tid] = (float)bi;
            }
            break;
        }

        // ---- Phase D: bulk head nonlinearities (keys/erase/add) ----
        for (int i = tid; i < BT * 1024; i += 256) {
            int row = i >> 10, o = i & 1023;
            float v = heads[row * HSTR + o];
            v = (o >= 512 && o < 768) ? sigf(v) : tanhf_fast(v); // erase: sigmoid, keys/add: tanh
            heads[row * HSTR + o] = v;
        }
        __syncthreads();

        // ---- Phase E: cosine dots + norms, 16 lanes per row, shfl reduce ----
        {
            const int b  = wv * 2 + hi;    // 8 waves x 2 half-waves = 16 rows
            const int ml = nn;             // lane's slice of the M dimension
            float rkv[16], wkv[16];
            float kr = 0.f, kw = 0.f;
            #pragma unroll
            for (int j = 0; j < 16; ++j) {
                rkv[j] = heads[b * HSTR + ml + j * 16];
                wkv[j] = heads[b * HSTR + kM + ml + j * 16];
                kr += rkv[j] * rkv[j];
                kw += wkv[j] * wkv[j];
            }
            for (int n = 0; n < kN; ++n) {
                const float* mrow = memws + ((size_t)(b0 + b) * kN + n) * kM;
                float dr_ = 0.f, dw_ = 0.f, mn2 = 0.f;
                #pragma unroll
                for (int j = 0; j < 16; ++j) {
                    float mv = mrow[ml + j * 16];
                    dr_ += rkv[j] * mv; dw_ += wkv[j] * mv; mn2 += mv * mv;
                }
                #pragma unroll
                for (int off = 1; off < 16; off <<= 1) {
                    dr_ += __shfl_xor(dr_, off, 32);
                    dw_ += __shfl_xor(dw_, off, 32);
                    mn2 += __shfl_xor(mn2, off, 32);
                }
                if (ml == 0) {
                    drs[b * kN + n] = dr_;
                    dws[b * kN + n] = dw_;
                    mns[b * kN + n] = sqrtf(mn2);
                }
            }
            #pragma unroll
            for (int off = 1; off < 16; off <<= 1) {
                kr += __shfl_xor(kr, off, 32);
                kw += __shfl_xor(kw, off, 32);
            }
            if (ml == 0) { knr[b] = sqrtf(kr); knw[b] = sqrtf(kw); }
        }
        __syncthreads();

        // ---- Phase F: per-row addressing (N=10, serial per batch row) ----
        if (tid < BT) {
            int b = tid;
            float rks  = softplusf(heads[b * HSTR + 1028]);
            float wks  = softplusf(heads[b * HSTR + 1029]);
            float rig  = sigf(heads[b * HSTR + 1030]);
            float wig  = sigf(heads[b * HSTR + 1031]);
            float rsh[3], wsh[3];
            {
                float a0 = heads[b * HSTR + 1032], a1 = heads[b * HSTR + 1033], a2 = heads[b * HSTR + 1034];
                float mx = fmaxf(a0, fmaxf(a1, a2));
                rsh[0] = __expf(a0 - mx); rsh[1] = __expf(a1 - mx); rsh[2] = __expf(a2 - mx);
                float s = rsh[0] + rsh[1] + rsh[2];
                rsh[0] /= s; rsh[1] /= s; rsh[2] /= s;
                a0 = heads[b * HSTR + 1035]; a1 = heads[b * HSTR + 1036]; a2 = heads[b * HSTR + 1037];
                mx = fmaxf(a0, fmaxf(a1, a2));
                wsh[0] = __expf(a0 - mx); wsh[1] = __expf(a1 - mx); wsh[2] = __expf(a2 - mx);
                s = wsh[0] + wsh[1] + wsh[2];
                wsh[0] /= s; wsh[1] /= s; wsh[2] /= s;
            }
            float rshp = softplusf(heads[b * HSTR + 1038]) + 1.f;
            float wshp = softplusf(heads[b * HSTR + 1039]) + 1.f;

            float wrnew[kN], wwnew[kN];
            address10(&drs[b * kN], fmaxf(knr[b], 1e-8f), &mns[b * kN], rks, rig, rsh, rshp,
                      &wr_ls[b * kN], t > 1, wrnew);
            address10(&dws[b * kN], fmaxf(knw[b], 1e-8f), &mns[b * kN], wks, wig, wsh, wshp,
                      &ww_ls[b * kN], t > 0, wwnew);
            #pragma unroll
            for (int n = 0; n < kN; ++n) {
                wn_ls[b * kN + n] = wrnew[n];
                if (t != 0) wr_ls[b * kN + n] = wrnew[n];   // carry update (t==0 keeps w_prev)
                ww_ls[b * kN + n] = wwnew[n];
            }
        }
        __syncthreads();

        // ---- Phase G: read_vec from pre-update memory, then erase/add update ----
        for (int i = tid; i < BT * kM; i += 256) {
            int b = i >> 8, m = i & 255;
            float* mrow = memws + (size_t)(b0 + b) * kN * kM + m;
            float er = heads[b * HSTR + 2 * kM + m];
            float ad = heads[b * HSTR + 3 * kM + m];
            float rv = 0.f;
            #pragma unroll
            for (int n = 0; n < kN; ++n) {
                float mv = mrow[n * kM];
                rv += mv * wn_ls[b * kN + n];
                float wn = ww_ls[b * kN + n];
                mrow[n * kM] = mv * (1.f - er * wn) + ad * wn;
            }
            rvec[i] = (t == 0) ? 0.f : rv;
        }
        __syncthreads();
    }
}

// ------------------------------- launcher ----------------------------------

extern "C" void kernel_launch(void* const* d_in, const int* in_sizes, int n_in,
                              void* d_out, int out_size, void* d_ws, size_t ws_size,
                              hipStream_t stream) {
    const float* z_seq = (const float*)d_in[0];
    const float* W_ih  = (const float*)d_in[1];
    const float* W_hh  = (const float*)d_in[2];
    const float* b_ih  = (const float*)d_in[3];
    const float* b_hh  = (const float*)d_in[4];
    HeadPtrs hp;
    for (int j = 0; j < 13; ++j) {
        hp.W[j] = (const float*)d_in[5 + 2 * j];
        hp.b[j] = (const float*)d_in[6 + 2 * j];
    }
    const float* memory_init = (const float*)d_in[31];
    const float* gamma       = (const float*)d_in[32];
    const float* beta        = (const float*)d_in[33];

    uint8_t* ws = (uint8_t*)d_ws;
    unsigned short* Wa    = (unsigned short*)(ws + OFF_WA);
    unsigned short* Wh    = (unsigned short*)(ws + OFF_WH);
    float*          bg    = (float*)(ws + OFF_BG);
    float*          bh    = (float*)(ws + OFF_BH);
    unsigned short* zn    = (unsigned short*)(ws + OFF_ZN);
    float*          memws = (float*)(ws + OFF_MEM);

    pack_gate_weights<<<(kG * kKA + 255) / 256, 256, 0, stream>>>(W_ih, W_hh, b_ih, b_hh, Wa, bg);
    pack_head_weights<<<(kHO * kH + 255) / 256, 256, 0, stream>>>(hp, Wh, bh);
    layernorm_cast<<<(kB * kZ + 255) / 256, 256, 0, stream>>>(z_seq, gamma, beta, zn);

    (void)hipFuncSetAttribute((const void*)ntm_persistent,
                              hipFuncAttributeMaxDynamicSharedMemorySize, SMEM_BYTES);
    ntm_persistent<<<dim3(kB / BT), dim3(256), SMEM_BYTES, stream>>>(
        Wa, Wh, bg, bh, zn, memws, memory_init, (float*)d_out);
}
